// TuckerLinear_2216203125285
// MI455X (gfx1250) — matrix operand order repaired and checked by reading, not compiled, not run on hardware
//
#include <hip/hip_runtime.h>
#include <hip/hip_bf16.h>

typedef __attribute__((ext_vector_type(2)))  float        v2f;
typedef __attribute__((ext_vector_type(8)))  float        v8f;
typedef __attribute__((ext_vector_type(4)))  float        v4f;
typedef __attribute__((ext_vector_type(4)))  unsigned int v4u;
typedef __attribute__((ext_vector_type(16))) __bf16       v16bf;

#define TILE 128
#define KT32 32     // bf16 GEMM k-slab
#define KT4  16     // f32 GEMM k-slab
#define LDSS 20     // padded LDS row stride in dwords (16B-alignable, conflict-light)

union FragU { v16bf v; v4u q[2]; };

// ---------------------------------------------------------------------------
// Kron-of-3-factors builder: Fout[n][q] = f0[n/256][q/64]*f1[(n/16)%16][(q/8)%8]*f2[n%16][q%8]
// ---------------------------------------------------------------------------
__global__ __launch_bounds__(256) void build_kron3(
    const float* __restrict__ f0, const float* __restrict__ f1,
    const float* __restrict__ f2, float* __restrict__ Fout)
{
    int idx = blockIdx.x * 256 + threadIdx.x;   // 0 .. 4096*512-1
    int q = idx & 511;
    int n = idx >> 9;
    Fout[idx] = f0[(n >> 8) * 8        + (q >> 6)]
              * f1[((n >> 4) & 15) * 8 + ((q >> 3) & 7)]
              * f2[(n & 15) * 8        + (q & 7)];
}

// fwd[inv[j]] = j  (invert the output permutation)
__global__ __launch_bounds__(256) void invert_perm(
    const int* __restrict__ inv, int* __restrict__ fwd)
{
    int j = blockIdx.x * 256 + threadIdx.x;
    fwd[inv[j]] = j;
}

// ---------------------------------------------------------------------------
// Exact f32 GEMM via V_WMMA_F32_16X16X4_F32: C[M,N] = A[M,K] @ B[N,K]^T.
// MODE 0: Cf[m*ldc+n] = acc
// MODE 1: j=rowmap[m], i=colmap[n]; v = acc*alpha*vec[j];
//         split v -> bf16 hi/lo planes Ch/Cl at [j*ldc+i].
// ---------------------------------------------------------------------------
template<int MODE>
__global__ __launch_bounds__(256) void gemm_tn_f32(
    const float* __restrict__ A, const float* __restrict__ Bm,
    float* __restrict__ Cf,
    unsigned short* __restrict__ Ch, unsigned short* __restrict__ Cl,
    int M, int N, int K, int ldc,
    const float* __restrict__ vec, const float* __restrict__ alpha,
    const int* __restrict__ colmap, const int* __restrict__ rowmap)
{
    __shared__ float As[TILE * LDSS];
    __shared__ float Bs[TILE * LDSS];

    const int tid  = threadIdx.x;
    const int bm   = blockIdx.y * TILE;
    const int bn   = blockIdx.x * TILE;
    const int wave = tid >> 5;
    const int lane = tid & 31;
    const int wm   = (wave & 3) * 32;
    const int wn   = (wave >> 2) * 64;
    const int lm   = lane & 15;
    const int kg   = lane >> 4;

    v8f acc[2][4];
    const v8f vzero = {0.f,0.f,0.f,0.f,0.f,0.f,0.f,0.f};
    #pragma unroll
    for (int t = 0; t < 2; ++t)
        #pragma unroll
        for (int u = 0; u < 4; ++u) acc[t][u] = vzero;

    for (int kt = 0; kt < K; kt += KT4) {
        __syncthreads();
        #pragma unroll
        for (int i = 0; i < 2; ++i) {
            int f   = tid + i * 256;
            int row = f >> 2;
            int c4  = (f & 3) << 2;
            v4f av = *(const v4f*)(A  + (size_t)(bm + row) * K + kt + c4);
            *(v4f*)&As[row * LDSS + c4] = av;
            v4f bv = *(const v4f*)(Bm + (size_t)(bn + row) * K + kt + c4);
            *(v4f*)&Bs[row * LDSS + c4] = bv;
        }
        __syncthreads();

        if (kt + KT4 < K) {
            int row = tid >> 1;
            int c8  = (tid & 1) << 3;
            __builtin_prefetch(&A [(size_t)(bm + row) * K + kt + KT4 + c8], 0, 1);
            __builtin_prefetch(&Bm[(size_t)(bn + row) * K + kt + KT4 + c8], 0, 1);
        }

        #pragma unroll
        for (int kk = 0; kk < KT4; kk += 4) {
            v2f a[2], b[4];
            #pragma unroll
            for (int t = 0; t < 2; ++t)
                a[t] = *(const v2f*)&As[(wm + t * 16 + lm) * LDSS + kk + kg * 2];
            #pragma unroll
            for (int u = 0; u < 4; ++u)
                b[u] = *(const v2f*)&Bs[(wn + u * 16 + lm) * LDSS + kk + kg * 2];
            #pragma unroll
            for (int t = 0; t < 2; ++t)
                #pragma unroll
                for (int u = 0; u < 4; ++u)
                    acc[t][u] = __builtin_amdgcn_wmma_f32_16x16x4_f32(
                        false, a[t], false, b[u], (short)0, acc[t][u], false, false);
        }
    }

    const float alp = (MODE == 1) ? alpha[0] : 0.f;
    #pragma unroll
    for (int t = 0; t < 2; ++t) {
        #pragma unroll
        for (int u = 0; u < 4; ++u) {
            #pragma unroll
            for (int v = 0; v < 8; ++v) {
                int m = bm + wm + t * 16 + v + 8 * kg;
                int n = bn + wn + u * 16 + lm;
                float val = acc[t][u][v];
                if (MODE == 0) {
                    Cf[(size_t)m * ldc + n] = val;
                } else {
                    int j = rowmap[m];
                    int i = colmap[n];
                    float sv = val * alp * vec[j];
                    unsigned int b  = __float_as_uint(sv);
                    float        hf = __uint_as_float(b & 0xFFFF0000u);
                    float        lf = sv - hf;
                    Ch[(size_t)j * ldc + i] = (unsigned short)(b >> 16);
                    Cl[(size_t)j * ldc + i] = (unsigned short)(__float_as_uint(lf) >> 16);
                }
            }
        }
    }
}

// ---------------------------------------------------------------------------
// Main GEMM, bf16x3 split-precision on V_WMMA_F32_16X16X32_BF16:
//   out[m,n] = sum_i x[m,i]*W'[n,i] + bias[n]
//   x split hi/lo on the fly during LDS staging; W' pre-split (Wh/Wl bf16 bits).
//   acc += xh*wh + xh*wl + xl*wh   (f32 accumulation, lo*lo dropped ~2^-32)
// ---------------------------------------------------------------------------
__global__ __launch_bounds__(256) void gemm_bf16x3(
    const float* __restrict__ X,
    const unsigned short* __restrict__ Wh, const unsigned short* __restrict__ Wl,
    float* __restrict__ C, int M, int N, int K,
    const float* __restrict__ bias)
{
    __shared__ unsigned int Ah[TILE * LDSS];
    __shared__ unsigned int Al[TILE * LDSS];
    __shared__ unsigned int Bh[TILE * LDSS];
    __shared__ unsigned int Bl[TILE * LDSS];

    const int tid  = threadIdx.x;
    const int bm   = blockIdx.y * TILE;
    const int bn   = blockIdx.x * TILE;
    const int wave = tid >> 5;
    const int lane = tid & 31;
    const int wm   = (wave & 3) * 32;
    const int wn   = (wave >> 2) * 64;
    const int lm   = lane & 15;
    const int kg   = lane >> 4;

    v8f acc[2][4];
    const v8f vzero = {0.f,0.f,0.f,0.f,0.f,0.f,0.f,0.f};
    #pragma unroll
    for (int t = 0; t < 2; ++t)
        #pragma unroll
        for (int u = 0; u < 4; ++u) acc[t][u] = vzero;

    const int srow = tid >> 1;       // staging row 0..127
    const int shlf = tid & 1;        // staging half (16 elems each)

    for (int kt = 0; kt < K; kt += KT32) {
        __syncthreads();
        // ---- stage x slab with on-the-fly hi/lo bf16 split (bit ops only) ----
        {
            const float* xr = X + (size_t)(bm + srow) * K + kt + shlf * 16;
            unsigned int* ah = &Ah[srow * LDSS + shlf * 8];
            unsigned int* al = &Al[srow * LDSS + shlf * 8];
            #pragma unroll
            for (int g = 0; g < 4; ++g) {
                v4f v = *(const v4f*)(xr + g * 4);
                unsigned int b0 = __float_as_uint(v.x), b1 = __float_as_uint(v.y);
                unsigned int b2 = __float_as_uint(v.z), b3 = __float_as_uint(v.w);
                float l0 = v.x - __uint_as_float(b0 & 0xFFFF0000u);
                float l1 = v.y - __uint_as_float(b1 & 0xFFFF0000u);
                float l2 = v.z - __uint_as_float(b2 & 0xFFFF0000u);
                float l3 = v.w - __uint_as_float(b3 & 0xFFFF0000u);
                ah[g * 2 + 0] = (b0 >> 16) | (b1 & 0xFFFF0000u);
                ah[g * 2 + 1] = (b2 >> 16) | (b3 & 0xFFFF0000u);
                al[g * 2 + 0] = (__float_as_uint(l0) >> 16) | (__float_as_uint(l1) & 0xFFFF0000u);
                al[g * 2 + 1] = (__float_as_uint(l2) >> 16) | (__float_as_uint(l3) & 0xFFFF0000u);
            }
            // ---- stage W' hi/lo slabs (raw bf16 bits, coalesced b128) ----
            const v4u* whp = (const v4u*)(Wh + (size_t)(bn + srow) * K + kt) + shlf * 2;
            const v4u* wlp = (const v4u*)(Wl + (size_t)(bn + srow) * K + kt) + shlf * 2;
            *(v4u*)&Bh[srow * LDSS + shlf * 8 + 0] = whp[0];
            *(v4u*)&Bh[srow * LDSS + shlf * 8 + 4] = whp[1];
            *(v4u*)&Bl[srow * LDSS + shlf * 8 + 0] = wlp[0];
            *(v4u*)&Bl[srow * LDSS + shlf * 8 + 4] = wlp[1];
        }
        __syncthreads();

        if (kt + KT32 < K) {
            __builtin_prefetch(&X [(size_t)(bm + srow) * K + kt + KT32 + shlf * 16], 0, 1);
            __builtin_prefetch(&Wh[(size_t)(bn + srow) * K + kt + KT32 + shlf * 16], 0, 1);
            __builtin_prefetch(&Wl[(size_t)(bn + srow) * K + kt + KT32 + shlf * 16], 0, 1);
        }

        // ---- fragments: A 16x32 bf16 layout => lane chunks [kg*8,+8) and [16+kg*8,+8)
        FragU fah[2], fal[2];
        #pragma unroll
        for (int t = 0; t < 2; ++t) {
            int base = (wm + t * 16 + lm) * LDSS;
            fah[t].q[0] = *(const v4u*)&Ah[base + kg * 4];
            fah[t].q[1] = *(const v4u*)&Ah[base + 8 + kg * 4];
            fal[t].q[0] = *(const v4u*)&Al[base + kg * 4];
            fal[t].q[1] = *(const v4u*)&Al[base + 8 + kg * 4];
        }
        // ---- B 32x16 bf16 layout => lane's column, chunks [kg*8,+8) and [16+kg*8,+8)
        #pragma unroll
        for (int u = 0; u < 4; ++u) {
            FragU fbh, fbl;
            int base = (wn + u * 16 + lm) * LDSS;
            fbh.q[0] = *(const v4u*)&Bh[base + kg * 4];
            fbh.q[1] = *(const v4u*)&Bh[base + 8 + kg * 4];
            fbl.q[0] = *(const v4u*)&Bl[base + kg * 4];
            fbl.q[1] = *(const v4u*)&Bl[base + 8 + kg * 4];
            #pragma unroll
            for (int t = 0; t < 2; ++t)
                acc[t][u] = __builtin_amdgcn_wmma_f32_16x16x32_bf16(
                    false, fah[t].v, false, fbh.v, (short)0, acc[t][u], false, false);
            #pragma unroll
            for (int t = 0; t < 2; ++t)
                acc[t][u] = __builtin_amdgcn_wmma_f32_16x16x32_bf16(
                    false, fah[t].v, false, fbl.v, (short)0, acc[t][u], false, false);
            #pragma unroll
            for (int t = 0; t < 2; ++t)
                acc[t][u] = __builtin_amdgcn_wmma_f32_16x16x32_bf16(
                    false, fal[t].v, false, fbh.v, (short)0, acc[t][u], false, false);
        }
    }

    #pragma unroll
    for (int t = 0; t < 2; ++t) {
        #pragma unroll
        for (int u = 0; u < 4; ++u) {
            #pragma unroll
            for (int v = 0; v < 8; ++v) {
                int m = bm + wm + t * 16 + v + 8 * kg;
                int n = bn + wn + u * 16 + lm;
                C[(size_t)m * N + n] = acc[t][u][v] + bias[n];
            }
        }
    }
}

// ---------------------------------------------------------------------------
// 0:x 1:core 2:fo0 3:fo1 4:fo2 5:fi0 6:fi1 7:fi2 8:bias 9:alpha
// 10:per_dim_scale 11:input_perm(i32) 12:output_inv_perm(i32)
// ---------------------------------------------------------------------------
extern "C" void kernel_launch(void* const* d_in, const int* in_sizes, int n_in,
                              void* d_out, int out_size, void* d_ws, size_t ws_size,
                              hipStream_t stream)
{
    const float* x     = (const float*)d_in[0];
    const float* core2 = (const float*)d_in[1];   // 8^6 C-order == [512,512] row-major
    const float* fo0   = (const float*)d_in[2];
    const float* fo1   = (const float*)d_in[3];
    const float* fo2   = (const float*)d_in[4];
    const float* fi0   = (const float*)d_in[5];
    const float* fi1   = (const float*)d_in[6];
    const float* fi2   = (const float*)d_in[7];
    const float* bias  = (const float*)d_in[8];
    const float* alpha = (const float*)d_in[9];
    const float* pds   = (const float*)d_in[10];
    const int* in_perm = (const int*)d_in[11];
    const int* out_inv = (const int*)d_in[12];
    float* out = (float*)d_out;

    // Workspace: FI 8MB | FO 8MB | Gt 8MB | Wph 32MB | Wpl 32MB | fwd 16KB
    float* ws = (float*)d_ws;
    float* FI = ws;
    float* FO = FI + (size_t)4096 * 512;
    float* Gt = FO + (size_t)4096 * 512;
    unsigned short* Wph = (unsigned short*)(Gt + (size_t)4096 * 512);
    unsigned short* Wpl = Wph + (size_t)4096 * 4096;
    int* fwd = (int*)(Wpl + (size_t)4096 * 4096);

    const int KRON_BLOCKS = (4096 * 512) / 256;
    build_kron3<<<KRON_BLOCKS, 256, 0, stream>>>(fi0, fi1, fi2, FI);
    build_kron3<<<KRON_BLOCKS, 256, 0, stream>>>(fo0, fo1, fo2, FO);
    invert_perm<<<4096 / 256, 256, 0, stream>>>(out_inv, fwd);

    // Gt[n,p] = sum_q FI[n,q] * core2[p,q]   (exact f32 WMMA)
    gemm_tn_f32<0><<<dim3(512 / TILE, 4096 / TILE), 256, 0, stream>>>(
        FI, core2, Gt, nullptr, nullptr, 4096, 512, 512, 512,
        nullptr, nullptr, nullptr, nullptr);

    // W[r,k] = sum_p FO[r,p]*Gt[k,p]; scatter+scale+split -> Wph/Wpl bf16 planes
    gemm_tn_f32<1><<<dim3(4096 / TILE, 4096 / TILE), 256, 0, stream>>>(
        FO, Gt, nullptr, Wph, Wpl, 4096, 4096, 512, 4096,
        pds, alpha, in_perm, fwd);

    // out = x @ W'^T + bias  via bf16x3 compensated WMMA
    gemm_bf16x3<<<dim3(4096 / TILE, 8192 / TILE), 256, 0, stream>>>(
        x, Wph, Wpl, out, 8192, 4096, 4096, bias);
}
